// GQAttention_60352880444037
// MI455X (gfx1250) — matrix operand order repaired and checked by reading, not compiled, not run on hardware
//
#include <hip/hip_runtime.h>
#include <hip/hip_bf16.h>

// ---------------------------------------------------------------------------
// GQA attention forward for MI455X (gfx1250), bf16 WMMA + f32 accumulate.
// B=2, S=2048, DIM=2048, 16 q-heads / 4 kv-heads, head_dim=128.
// ---------------------------------------------------------------------------

#define B_       2
#define S_       2048
#define DIM_     2048
#define NH_      16
#define NKV_     4
#define HD_      128
#define GRP_     (NH_ / NKV_)     // 4
#define M_       (B_ * S_)        // 4096
#define KVD_     (NKV_ * HD_)     // 512

typedef __attribute__((ext_vector_type(16))) __bf16 v16bf;
typedef __attribute__((ext_vector_type(8)))  __bf16 v8bf;
typedef __attribute__((ext_vector_type(8)))  float  v8f;

__device__ __forceinline__ __bf16 f2bf(float f) {
    unsigned u = __builtin_bit_cast(unsigned, f);
    unsigned r = u + 0x7FFFu + ((u >> 16) & 1u);      // round-to-nearest-even
    unsigned short h = (unsigned short)(r >> 16);
    return __builtin_bit_cast(__bf16, h);
}
__device__ __forceinline__ float bf2f(__bf16 b) {
    unsigned u = ((unsigned)__builtin_bit_cast(unsigned short, b)) << 16;
    return __builtin_bit_cast(float, u);
}

// A-fragment (16x32 bf16), ISA 7.12.2 layout:
//   lane 0-15 : row = lane,    K 0-7 in v0-3,  K 16-23 in v4-7
//   lane 16-31: row = lane-16, K 8-15 in v0-3, K 24-31 in v4-7
__device__ __forceinline__ v16bf load_a_frag(const __bf16* base, int row_stride, int lane) {
    const __bf16* p = base + (size_t)(lane & 15) * row_stride + ((lane >> 4) << 3);
    union { v16bf v; v8bf h[2]; } u;
    u.h[0] = *(const v8bf*)(p);
    u.h[1] = *(const v8bf*)(p + 16);
    return u.v;
}

// B-fragment (32x16 bf16): lane holds column (lane&15);
//   lanes 0-15 hold K 0-7, 16-23; lanes 16-31 hold K 8-15, 24-31 (same as A).
__device__ __forceinline__ v16bf load_b_frag(const __bf16* base, int col_stride, int lane) {
    const __bf16* p = base + (size_t)(lane & 15) * col_stride + ((lane >> 4) << 3);
    union { v16bf v; v8bf h[2]; } u;
    u.h[0] = *(const v8bf*)(p);
    u.h[1] = *(const v8bf*)(p + 16);
    return u.v;
}

// ---------------------------------------------------------------------------
// f32 -> bf16 conversion (grid-stride)
// ---------------------------------------------------------------------------
__global__ void cvt_f32_bf16(const float* __restrict__ s, __bf16* __restrict__ d, int n) {
    for (int i = blockIdx.x * blockDim.x + threadIdx.x; i < n; i += gridDim.x * blockDim.x)
        d[i] = f2bf(s[i]);
}

// ---------------------------------------------------------------------------
// GEMM: C[M,N] = A[M,K] * Bt[N,K]^T   (both operands K-contiguous)
// block tile 256x128, 8 waves (4M x 2N), wave tile 64x64, k-step 32.
// Per k-step: 16 WMMAs; each B fragment feeds 4 WMMAs (~32 FLOP/B from L2).
// ---------------------------------------------------------------------------
template <int OUT_F32>
__global__ __launch_bounds__(256)
void gemm_bf16_wmma(const __bf16* __restrict__ A, const __bf16* __restrict__ Bt,
                    void* __restrict__ Cout, int M, int N, int K) {
    const int lane = threadIdx.x & 31;
    const int wave = threadIdx.x >> 5;
    const int wm = blockIdx.x * 256 + (wave & 3) * 64;
    const int wn = blockIdx.y * 128 + (wave >> 2) * 64;

    v8f acc[4][4];
#pragma unroll
    for (int i = 0; i < 4; ++i)
#pragma unroll
        for (int j = 0; j < 4; ++j) acc[i][j] = (v8f)(0.0f);

    for (int k0 = 0; k0 < K; k0 += 32) {
        // prefetch next k-tile (global_prefetch_b8; 192MB L2 streaming)
        if (k0 + 32 < K) {
            __builtin_prefetch(A + (size_t)(wm + (lane & 15)) * K + k0 + 32, 0, 1);
            __builtin_prefetch(Bt + (size_t)(wn + (lane & 15)) * K + k0 + 32, 0, 1);
        }
        v16bf a[4];
#pragma unroll
        for (int i = 0; i < 4; ++i)
            a[i] = load_a_frag(A + (size_t)(wm + i * 16) * K + k0, K, lane);
#pragma unroll
        for (int j = 0; j < 4; ++j) {
            v16bf b = load_b_frag(Bt + (size_t)(wn + j * 16) * K + k0, K, lane);
#pragma unroll
            for (int i = 0; i < 4; ++i)
                acc[i][j] = __builtin_amdgcn_wmma_f32_16x16x32_bf16(
                    false, a[i], false, b, (short)0, acc[i][j], false, false);
        }
    }

    const int row0 = wm + (lane >> 4) * 8;     // C layout: VGPR r -> row r / r+8
    const int col0 = wn + (lane & 15);
#pragma unroll
    for (int i = 0; i < 4; ++i)
#pragma unroll
        for (int j = 0; j < 4; ++j)
#pragma unroll
            for (int r = 0; r < 8; ++r) {
                size_t idx = (size_t)(row0 + i * 16 + r) * N + (col0 + j * 16);
                float v = acc[i][j][r];
                if (OUT_F32) ((float*)Cout)[idx] = v;
                else         ((__bf16*)Cout)[idx] = f2bf(v);
            }
}

// ---------------------------------------------------------------------------
// RoPE in place on bf16 tensor shaped (B, S, heads, 128)
// ---------------------------------------------------------------------------
__global__ void rope_kernel(__bf16* __restrict__ X, const float* __restrict__ cs,
                            const float* __restrict__ sn, int heads, int total) {
    int idx = blockIdx.x * blockDim.x + threadIdx.x;
    if (idx >= total) return;
    int d = idx & 63;
    int t = idx >> 6;
    int h = t % heads; t /= heads;
    int s = t % S_;
    int b = t / S_;
    size_t base = (((size_t)(b * S_ + s)) * heads + h) * HD_;
    float x1 = bf2f(X[base + d]);
    float x2 = bf2f(X[base + 64 + d]);
    float c = cs[s * 64 + d], sv = sn[s * 64 + d];
    X[base + d]      = f2bf(x1 * c - x2 * sv);
    X[base + 64 + d] = f2bf(x1 * sv + x2 * c);
}

// ---------------------------------------------------------------------------
// Flash attention: one block = (b, head, 128 q rows); wave owns 16 q rows.
// Key tiles of 32 with online softmax; P re-laid out via LDS; V transposed
// into LDS cooperatively so P*V B-fragments are contiguous.
// ---------------------------------------------------------------------------
__global__ __launch_bounds__(256)
void flash_attn(const __bf16* __restrict__ Q, const __bf16* __restrict__ Kt,
                const __bf16* __restrict__ V, const unsigned char* __restrict__ mask,
                __bf16* __restrict__ O) {
    const int lane = threadIdx.x & 31;
    const int wave = threadIdx.x >> 5;
    const int qb   = blockIdx.x;          // q row block (0..15)
    const int h    = blockIdx.y;          // q head
    const int b    = blockIdx.z;
    const int kvh  = h / GRP_;

    __shared__ __bf16 Vt[HD_][40];        // transposed V tile: [dim][key 0..31]
    __shared__ __bf16 Pld[8][16][40];     // per-wave probability tile

    const float sc = 1.44269504088896f / 11.3137084989848f;  // log2e / sqrt(128)

    const __bf16* Qbase = Q + ((size_t)b * S_) * (NH_ * HD_) + (size_t)h * HD_;
    const __bf16* Kbase = Kt + ((size_t)b * S_) * KVD_ + (size_t)kvh * HD_;
    const __bf16* Vbase = V  + ((size_t)b * S_) * KVD_ + (size_t)kvh * HD_;
    const unsigned char* mb = mask + (size_t)b * S_;

    const int qr0 = qb * 128 + wave * 16;

    v16bf qf[4];
#pragma unroll
    for (int t = 0; t < 4; ++t)
        qf[t] = load_a_frag(Qbase + (size_t)qr0 * (NH_ * HD_) + t * 32, NH_ * HD_, lane);

    v8f acc[8];
#pragma unroll
    for (int t = 0; t < 8; ++t) acc[t] = (v8f)(0.0f);
    float m[8], l[8];
#pragma unroll
    for (int r = 0; r < 8; ++r) { m[r] = -1e30f; l[r] = 0.0f; }

    for (int j0 = 0; j0 < S_; j0 += 32) {
        // prefetch next key tile of K and V
        if (j0 + 32 < S_) {
            __builtin_prefetch(Kbase + (size_t)(j0 + 32 + (lane & 15)) * KVD_, 0, 1);
            __builtin_prefetch(Vbase + (size_t)(j0 + 32 + (lane & 15)) * KVD_, 0, 1);
        }
        // cooperative transposed V tile load: 32 keys x 128 dims
        __syncthreads();
        {
            int key  = (int)(threadIdx.x >> 3);          // 0..31
            int dseg = ((int)threadIdx.x & 7) * 16;      // 0..112
            const __bf16* vp = Vbase + (size_t)(j0 + key) * KVD_ + dseg;
#pragma unroll
            for (int e = 0; e < 16; ++e) Vt[dseg + e][key] = vp[e];
        }
        __syncthreads();

        // scores: two 16x16 tiles (keys j0.. , j0+16..), K-dim = head_dim 128
        v8f c1 = (v8f)(0.0f), c2 = (v8f)(0.0f);
#pragma unroll
        for (int t = 0; t < 4; ++t) {
            v16bf bk0 = load_b_frag(Kbase + (size_t)j0 * KVD_ + t * 32, KVD_, lane);
            v16bf bk1 = load_b_frag(Kbase + (size_t)(j0 + 16) * KVD_ + t * 32, KVD_, lane);
            c1 = __builtin_amdgcn_wmma_f32_16x16x32_bf16(false, qf[t], false, bk0,
                                                         (short)0, c1, false, false);
            c2 = __builtin_amdgcn_wmma_f32_16x16x32_bf16(false, qf[t], false, bk1,
                                                         (short)0, c2, false, false);
        }

        // mask + scale (pre-multiplied by log2e), online softmax update
        const bool mA = mb[j0 + (lane & 15)] != 0;
        const bool mB = mb[j0 + 16 + (lane & 15)] != 0;
        float t1[8], t2[8];
#pragma unroll
        for (int r = 0; r < 8; ++r) {
            t1[r] = mA ? -1e30f : c1[r] * sc;
            t2[r] = mB ? -1e30f : c2[r] * sc;
        }
        float p1[8], p2[8];
#pragma unroll
        for (int r = 0; r < 8; ++r) {
            float v = fmaxf(t1[r], t2[r]);
#pragma unroll
            for (int off = 1; off < 16; off <<= 1) v = fmaxf(v, __shfl_xor(v, off, 32));
            float mnew = fmaxf(m[r], v);
            float alpha = exp2f(m[r] - mnew);
            p1[r] = exp2f(t1[r] - mnew);
            p2[r] = exp2f(t2[r] - mnew);
            float rs = p1[r] + p2[r];
#pragma unroll
            for (int off = 1; off < 16; off <<= 1) rs += __shfl_xor(rs, off, 32);
            l[r] = l[r] * alpha + rs;
            m[r] = mnew;
#pragma unroll
            for (int t = 0; t < 8; ++t) acc[t][r] *= alpha;
        }

        // C-layout P -> LDS (row, key) -> A-fragment
        const int rbase = (lane >> 4) * 8;
#pragma unroll
        for (int r = 0; r < 8; ++r) {
            Pld[wave][rbase + r][lane & 15]        = f2bf(p1[r]);
            Pld[wave][rbase + r][16 + (lane & 15)] = f2bf(p2[r]);
        }
        v16bf pf = load_a_frag(&Pld[wave][0][0], 40, lane);

        // O += P * V : contraction over 32 keys, 8 dim tiles of 16
#pragma unroll
        for (int t = 0; t < 8; ++t) {
            v16bf bv = load_b_frag(&Vt[t * 16][0], 40, lane);
            acc[t] = __builtin_amdgcn_wmma_f32_16x16x32_bf16(false, pf, false, bv,
                                                             (short)0, acc[t], false, false);
        }
    }

    // normalize and write (b, s, h, d) bf16
    const int rbase = (lane >> 4) * 8;
#pragma unroll
    for (int r = 0; r < 8; ++r) {
        float inv = 1.0f / l[r];
        int row = qr0 + rbase + r;
        size_t obase = (((size_t)(b * S_ + row)) * NH_ + h) * HD_;
#pragma unroll
        for (int t = 0; t < 8; ++t)
            O[obase + t * 16 + (lane & 15)] = f2bf(acc[t][r] * inv);
    }
}

// ---------------------------------------------------------------------------
// host launcher
// ---------------------------------------------------------------------------
extern "C" void kernel_launch(void* const* d_in, const int* in_sizes, int n_in,
                              void* d_out, int out_size, void* d_ws, size_t ws_size,
                              hipStream_t stream) {
    const float* x    = (const float*)d_in[0];
    const float* cosT = (const float*)d_in[1];
    const float* sinT = (const float*)d_in[2];
    const unsigned char* mask = (const unsigned char*)d_in[3];
    const float* Wq = (const float*)d_in[4];
    const float* Wk = (const float*)d_in[5];
    const float* Wv = (const float*)d_in[6];
    const float* Wo = (const float*)d_in[7];

    char* ws = (char*)d_ws;
    size_t off = 0;
    auto alloc = [&](size_t bytes) {
        size_t o = off;
        off = (off + bytes + 255) & ~(size_t)255;
        return o;
    };
    __bf16* xb  = (__bf16*)(ws + alloc((size_t)M_ * DIM_ * 2));
    __bf16* wqb = (__bf16*)(ws + alloc((size_t)DIM_ * DIM_ * 2));
    __bf16* wkb = (__bf16*)(ws + alloc((size_t)KVD_ * DIM_ * 2));
    __bf16* wvb = (__bf16*)(ws + alloc((size_t)KVD_ * DIM_ * 2));
    __bf16* wob = (__bf16*)(ws + alloc((size_t)DIM_ * DIM_ * 2));
    __bf16* Qb  = (__bf16*)(ws + alloc((size_t)M_ * DIM_ * 2));
    __bf16* Kb  = (__bf16*)(ws + alloc((size_t)M_ * KVD_ * 2));
    __bf16* Vb  = (__bf16*)(ws + alloc((size_t)M_ * KVD_ * 2));
    __bf16* Ab  = (__bf16*)(ws + alloc((size_t)M_ * DIM_ * 2));

    const int CT = 256, CG = 4096;
    cvt_f32_bf16<<<CG, CT, 0, stream>>>(x,  xb,  M_ * DIM_);
    cvt_f32_bf16<<<CG, CT, 0, stream>>>(Wq, wqb, DIM_ * DIM_);
    cvt_f32_bf16<<<CG, CT, 0, stream>>>(Wk, wkb, KVD_ * DIM_);
    cvt_f32_bf16<<<CG, CT, 0, stream>>>(Wv, wvb, KVD_ * DIM_);
    cvt_f32_bf16<<<CG, CT, 0, stream>>>(Wo, wob, DIM_ * DIM_);

    // projections (block tile 256x128)
    gemm_bf16_wmma<0><<<dim3(M_ / 256, DIM_ / 128), 256, 0, stream>>>(xb, wqb, Qb, M_, DIM_, DIM_);
    gemm_bf16_wmma<0><<<dim3(M_ / 256, KVD_ / 128), 256, 0, stream>>>(xb, wkb, Kb, M_, KVD_, DIM_);
    gemm_bf16_wmma<0><<<dim3(M_ / 256, KVD_ / 128), 256, 0, stream>>>(xb, wvb, Vb, M_, KVD_, DIM_);

    // RoPE
    {
        int totQ = B_ * S_ * NH_ * 64;
        int totK = B_ * S_ * NKV_ * 64;
        rope_kernel<<<(totQ + 255) / 256, 256, 0, stream>>>(Qb, cosT, sinT, NH_, totQ);
        rope_kernel<<<(totK + 255) / 256, 256, 0, stream>>>(Kb, cosT, sinT, NKV_, totK);
    }

    // attention
    flash_attn<<<dim3(S_ / 128, NH_, B_), 256, 0, stream>>>(Qb, Kb, Vb, mask, Ab);

    // output projection -> f32 d_out
    gemm_bf16_wmma<1><<<dim3(M_ / 256, DIM_ / 128), 256, 0, stream>>>(Ab, wob, (float*)d_out,
                                                                      M_, DIM_, DIM_);
}
